// MatryoshkaSAE_30073361007307
// MI455X (gfx1250) — matrix operand order repaired and checked by reading, not compiled, not run on hardware
//
#include <hip/hip_runtime.h>
#include <hip/hip_bf16.h>

// ---------------- types ----------------
typedef __bf16 bf16_t;
typedef __attribute__((ext_vector_type(8)))  bf16_t v8bf;
typedef __attribute__((ext_vector_type(16))) bf16_t v16bf;
typedef __attribute__((ext_vector_type(8)))  float  v8f;

#define BB 2048
#define DD 1024
#define SS 30720
#define N_TOTAL ((size_t)BB * (size_t)SS)
#define TOPK_TOTAL  (64u * 2048u)          // 131072
#define KAUX_TOTAL  (64u * 16u * 2048u)    // 2097152

// ---------------- WMMA fragment loaders (wave32; A and B both use this code's contiguous k = 16*(lane/16) + s order) ----------------
// A 16x32 bf16: lanes 0-15 -> M=lane, K=klo..klo+15 klo=0 ; lanes 16-31 same M, klo=16
__device__ __forceinline__ v16bf load_a_bf16(const bf16_t* __restrict__ A, size_t lda,
                                             int row, int k0, int lane) {
  const int klo = k0 + ((lane & 16) ? 16 : 0);
  const bf16_t* p = A + (size_t)row * lda + klo;
  v8bf lo = *(const v8bf*)p;
  v8bf hi = *(const v8bf*)(p + 8);
  v16bf f;
#pragma unroll
  for (int i = 0; i < 8; ++i) { f[i] = lo[i]; f[8 + i] = hi[i]; }
  return f;
}

// B 32x16 bf16 from N-major (transposed) layout: lane holds column N, 16 contiguous K values
__device__ __forceinline__ v16bf load_b_bf16(const bf16_t* __restrict__ Bt, size_t ldb,
                                             int col, int k0, int lane) {
  const int kk = k0 + ((lane & 16) ? 16 : 0);
  return *(const v16bf*)(Bt + (size_t)col * ldb + kk);
}

__device__ __forceinline__ v8f wmma_bf16(const v16bf& a, const v16bf& b, const v8f& c) {
  return __builtin_amdgcn_wmma_f32_16x16x32_bf16(false, a, false, b, (short)0, c, false, false);
}

__device__ __forceinline__ void load_frags(const bf16_t* __restrict__ A, size_t lda,
                                           const bf16_t* __restrict__ Bt, size_t ldb,
                                           int m_base, int n_base, int k0, int lane,
                                           v16bf a[4], v16bf b[4]) {
#pragma unroll
  for (int mi = 0; mi < 4; ++mi)
    a[mi] = load_a_bf16(A, lda, m_base + 16 * mi + (lane & 15), k0, lane);
#pragma unroll
  for (int ni = 0; ni < 4; ++ni)
    b[ni] = load_b_bf16(Bt, ldb, n_base + 16 * ni + (lane & 15), k0, lane);
}

__device__ __forceinline__ void do_wmma(const v16bf a[4], const v16bf b[4], v8f acc[4][4]) {
#pragma unroll
  for (int mi = 0; mi < 4; ++mi)
#pragma unroll
    for (int ni = 0; ni < 4; ++ni)
      acc[mi][ni] = wmma_bf16(a[mi], b[ni], acc[mi][ni]);
}

// Double-buffered K-loop: (kend-kbeg)/32 is even (>=2) for every call site.
__device__ __forceinline__ void gemm_mainloop(const bf16_t* __restrict__ A, size_t lda,
                                              const bf16_t* __restrict__ Bt, size_t ldb,
                                              int m_base, int n_base, int kbeg, int kend,
                                              int lane, v8f acc[4][4]) {
  v16bf a0[4], b0[4], a1[4], b1[4];
  load_frags(A, lda, Bt, ldb, m_base, n_base, kbeg, lane, a0, b0);
  int k0 = kbeg;
  for (; k0 < kend - 64; k0 += 64) {
    load_frags(A, lda, Bt, ldb, m_base, n_base, k0 + 32, lane, a1, b1);
    do_wmma(a0, b0, acc);
    load_frags(A, lda, Bt, ldb, m_base, n_base, k0 + 64, lane, a0, b0);
    do_wmma(a1, b1, acc);
  }
  load_frags(A, lda, Bt, ldb, m_base, n_base, k0 + 32, lane, a1, b1);
  do_wmma(a0, b0, acc);
  do_wmma(a1, b1, acc);
}

// ---------------- prep kernels ----------------
__global__ void k_zero_u32(unsigned* p, int n) {
  for (int i = blockIdx.x * blockDim.x + threadIdx.x; i < n; i += gridDim.x * blockDim.x) p[i] = 0u;
}

// tiled transpose + f32->bf16 : in [R,C] f32 -> out [C,R] bf16
__global__ __launch_bounds__(256) void k_transpose_cvt(const float* __restrict__ in,
                                                       bf16_t* __restrict__ out, int R, int C) {
  __shared__ float tile[32][33];
  const int tx = threadIdx.x & 31, ty = threadIdx.x >> 5;
  const int c0 = blockIdx.x * 32, r0 = blockIdx.y * 32;
#pragma unroll
  for (int i = 0; i < 32; i += 8)
    tile[ty + i][tx] = in[(size_t)(r0 + ty + i) * C + (c0 + tx)];
  __syncthreads();
#pragma unroll
  for (int i = 0; i < 32; i += 8)
    out[(size_t)(c0 + ty + i) * R + (r0 + tx)] = (bf16_t)tile[tx][ty + i];
}

__global__ void k_xc(const float* __restrict__ x, const float* __restrict__ b_dec,
                     bf16_t* __restrict__ xc) {
  int i = blockIdx.x * blockDim.x + threadIdx.x;
  if (i < BB * DD) xc[i] = (bf16_t)(x[i] - b_dec[i % DD]);
}

__global__ void k_dead(const int* __restrict__ nba, unsigned char* __restrict__ dead,
                       unsigned* any_dead) {
  int i = blockIdx.x * blockDim.x + threadIdx.x;
  if (i < SS) {
    unsigned char d = (nba[i] >= 20) ? 1 : 0;
    dead[i] = d;
    if (d) atomicOr(any_dead, 1u);
  }
}

__global__ void k_init_recon(const float* __restrict__ b_dec, float* __restrict__ recon) {
  int i = blockIdx.x * blockDim.x + threadIdx.x;
  if (i < BB * DD) recon[i] = b_dec[i % DD];
}

// ---------------- encoder GEMM: acts = relu(xc @ Wenc_t^T + b_enc) ----------------
__global__ __launch_bounds__(256, 1)
void k_encoder(const bf16_t* __restrict__ Xc, const bf16_t* __restrict__ Wt,
               const float* __restrict__ b_enc, float* __restrict__ acts) {
  const int lane = threadIdx.x & 31;
  const int wave = threadIdx.x >> 5;
  const int n_base = blockIdx.x * 512 + wave * 64;
  const int m_base = blockIdx.y * 64;
  const v8f vz = {0.f, 0.f, 0.f, 0.f, 0.f, 0.f, 0.f, 0.f};
  v8f acc[4][4];
#pragma unroll
  for (int mi = 0; mi < 4; ++mi)
#pragma unroll
    for (int ni = 0; ni < 4; ++ni) acc[mi][ni] = vz;

  gemm_mainloop(Xc, DD, Wt, DD, m_base, n_base, 0, DD, lane, acc);

  const int r0 = (lane & 16) ? 8 : 0;
  const int cl = lane & 15;
#pragma unroll
  for (int ni = 0; ni < 4; ++ni) {
    const int col = n_base + 16 * ni + cl;
    const float bias = b_enc[col];
#pragma unroll
    for (int mi = 0; mi < 4; ++mi)
#pragma unroll
      for (int r = 0; r < 8; ++r) {
        float v = acc[mi][ni][r] + bias;
        v = v > 0.f ? v : 0.f;
        // acts is a 252MB stream consumed by later passes: keep it out of the
        // weight-resident L2 working set (TH=NT).
        __builtin_nontemporal_store(v, &acts[(size_t)(m_base + 16 * mi + r0 + r) * SS + col]);
      }
  }
}

// ---------------- radix-select (2-level, 4096 buckets on float bits; acts >= 0) ----------------
__global__ void k_hist1(const float* __restrict__ acts, size_t n,
                        const unsigned char* __restrict__ mask, unsigned* __restrict__ counts) {
  __shared__ unsigned h[4096];
  for (int i = threadIdx.x; i < 4096; i += blockDim.x) h[i] = 0;
  __syncthreads();
  const size_t stride = (size_t)gridDim.x * blockDim.x;
  for (size_t i = (size_t)blockIdx.x * blockDim.x + threadIdx.x; i < n; i += stride) {
    if (mask) { int s = (int)(i % (size_t)SS); if (!mask[s]) continue; }
    float v = __builtin_nontemporal_load(&acts[i]);
    unsigned b = __float_as_uint(v) >> 20;
    atomicAdd(&h[b], 1u);
  }
  __syncthreads();
  for (int i = threadIdx.x; i < 4096; i += blockDim.x)
    if (h[i]) atomicAdd(&counts[i], h[i]);
}

__global__ void k_pick1(const unsigned* __restrict__ counts, unsigned k, unsigned* __restrict__ sel) {
  if (threadIdx.x == 0) {
    unsigned long long cum = 0;
    int bq = -1; unsigned rem = 0;
    for (int i = 4095; i >= 0; --i) {
      unsigned long long c = cum + counts[i];
      if (c >= (unsigned long long)k) { bq = i; rem = (unsigned)((unsigned long long)k - cum); break; }
      cum = c;
    }
    if (bq < 0) { sel[0] = 0; sel[1] = 0; sel[2] = 0; }
    else        { sel[0] = (unsigned)bq; sel[1] = rem; sel[2] = 1; }
  }
}

__global__ void k_hist2(const float* __restrict__ acts, size_t n,
                        const unsigned char* __restrict__ mask,
                        const unsigned* __restrict__ sel, unsigned* __restrict__ counts) {
  if (!sel[2]) return;  // uniform across grid
  const unsigned bq = sel[0];
  __shared__ unsigned h[4096];
  for (int i = threadIdx.x; i < 4096; i += blockDim.x) h[i] = 0;
  __syncthreads();
  const size_t stride = (size_t)gridDim.x * blockDim.x;
  for (size_t i = (size_t)blockIdx.x * blockDim.x + threadIdx.x; i < n; i += stride) {
    if (mask) { int s = (int)(i % (size_t)SS); if (!mask[s]) continue; }
    unsigned bits = __float_as_uint(__builtin_nontemporal_load(&acts[i]));
    if ((bits >> 20) == bq) atomicAdd(&h[(bits >> 8) & 0xFFFu], 1u);
  }
  __syncthreads();
  for (int i = threadIdx.x; i < 4096; i += blockDim.x)
    if (h[i]) atomicAdd(&counts[i], h[i]);
}

__global__ void k_pick2(const unsigned* __restrict__ counts, const unsigned* __restrict__ sel,
                        float* __restrict__ thr) {
  if (threadIdx.x == 0) {
    if (!sel[2]) { *thr = 0.f; return; }
    unsigned rem = sel[1];
    unsigned long long cum = 0;
    int sb = 0;
    for (int i = 4095; i >= 0; --i) { cum += counts[i]; if (cum >= rem) { sb = i; break; } }
    *thr = __uint_as_float((sel[0] << 20) | ((unsigned)sb << 8));
  }
}

// ---------------- scatter top-k + l1/l0 sums ----------------
__global__ void k_scatter_topk(const float* __restrict__ acts, const float* __restrict__ thrp,
                               float* __restrict__ out_topk, bf16_t* __restrict__ topk_bf,
                               size_t n, float* __restrict__ fscal) {
  const float thr = *thrp;
  float l1 = 0.f, cnt = 0.f;
  const size_t stride = (size_t)gridDim.x * blockDim.x;
  for (size_t i = (size_t)blockIdx.x * blockDim.x + threadIdx.x; i < n; i += stride) {
    float v = __builtin_nontemporal_load(&acts[i]);
    bool sel = (v >= thr) && (v > 0.f);
    float w = sel ? v : 0.f;
    __builtin_nontemporal_store(w, &out_topk[i]);
    __builtin_nontemporal_store((bf16_t)w, &topk_bf[i]);
    l1 += w;
    cnt += sel ? 1.f : 0.f;
  }
  __shared__ float sa[256], sb[256];
  int t = threadIdx.x;
  sa[t] = l1; sb[t] = cnt;
  __syncthreads();
  for (int s = 128; s > 0; s >>= 1) {
    if (t < s) { sa[t] += sa[t + s]; sb[t] += sb[t + s]; }
    __syncthreads();
  }
  if (t == 0) { atomicAdd(&fscal[0], sa[0]); atomicAdd(&fscal[1], sb[0]); }
}

__global__ void k_scatter_aux(const float* __restrict__ acts, const float* __restrict__ thrp,
                              const unsigned char* __restrict__ dead, bf16_t* __restrict__ aux_bf,
                              size_t n) {
  const float thr = *thrp;
  const size_t stride = (size_t)gridDim.x * blockDim.x;
  for (size_t i = (size_t)blockIdx.x * blockDim.x + threadIdx.x; i < n; i += stride) {
    int s = (int)(i % (size_t)SS);
    float v = __builtin_nontemporal_load(&acts[i]) - thr;
    float w = (dead[s] && v > 0.f) ? v : 0.f;
    __builtin_nontemporal_store((bf16_t)w, &aux_bf[i]);
  }
}

// ---------------- group decode GEMM: recon += acts_topk[:,kbeg:kend] @ W_dec[kbeg:kend,:] ----------------
__global__ __launch_bounds__(256, 1)
void k_decode(const bf16_t* __restrict__ A, const bf16_t* __restrict__ Bt,
              float* __restrict__ recon, int kbeg, int kend) {
  const int lane = threadIdx.x & 31;
  const int wave = threadIdx.x >> 5;
  const int n_base = blockIdx.x * 512 + wave * 64;
  const int m_base = blockIdx.y * 64;
  const int r0 = (lane & 16) ? 8 : 0;
  const int cl = lane & 15;
  v8f acc[4][4];
#pragma unroll
  for (int mi = 0; mi < 4; ++mi)
#pragma unroll
    for (int ni = 0; ni < 4; ++ni)
#pragma unroll
      for (int r = 0; r < 8; ++r)
        acc[mi][ni][r] = recon[(size_t)(m_base + 16 * mi + r0 + r) * DD + (n_base + 16 * ni + cl)];

  gemm_mainloop(A, SS, Bt, SS, m_base, n_base, kbeg, kend, lane, acc);

#pragma unroll
  for (int mi = 0; mi < 4; ++mi)
#pragma unroll
    for (int ni = 0; ni < 4; ++ni)
#pragma unroll
      for (int r = 0; r < 8; ++r)
        recon[(size_t)(m_base + 16 * mi + r0 + r) * DD + (n_base + 16 * ni + cl)] = acc[mi][ni][r];
}

// ---------------- aux GEMM: sum((acts_aux @ W_dec + recon - x)^2) ----------------
__global__ __launch_bounds__(256, 1)
void k_aux_gemm(const bf16_t* __restrict__ A, const bf16_t* __restrict__ Bt,
                const float* __restrict__ recon, const float* __restrict__ x,
                float* __restrict__ out_sum) {
  const int lane = threadIdx.x & 31;
  const int wave = threadIdx.x >> 5;
  const int n_base = blockIdx.x * 512 + wave * 64;
  const int m_base = blockIdx.y * 64;
  const int r0 = (lane & 16) ? 8 : 0;
  const int cl = lane & 15;
  v8f acc[4][4];
#pragma unroll
  for (int mi = 0; mi < 4; ++mi)
#pragma unroll
    for (int ni = 0; ni < 4; ++ni)
#pragma unroll
      for (int r = 0; r < 8; ++r) {
        size_t idx = (size_t)(m_base + 16 * mi + r0 + r) * DD + (n_base + 16 * ni + cl);
        acc[mi][ni][r] = recon[idx] - x[idx];  // C = -(residual): final D = recon_aux - residual
      }

  gemm_mainloop(A, SS, Bt, SS, m_base, n_base, 0, SS, lane, acc);

  float local = 0.f;
#pragma unroll
  for (int mi = 0; mi < 4; ++mi)
#pragma unroll
    for (int ni = 0; ni < 4; ++ni)
#pragma unroll
      for (int r = 0; r < 8; ++r) { float d = acc[mi][ni][r]; local += d * d; }
  atomicAdd(out_sum, local);
}

// ---------------- reductions / finalize ----------------
__global__ void k_sqerr(const float* __restrict__ a, const float* __restrict__ b, int n,
                        float* __restrict__ out) {
  float s = 0.f;
  for (int i = blockIdx.x * blockDim.x + threadIdx.x; i < n; i += gridDim.x * blockDim.x) {
    float d = a[i] - b[i];
    s += d * d;
  }
  __shared__ float sm[256];
  sm[threadIdx.x] = s;
  __syncthreads();
  for (int st = 128; st > 0; st >>= 1) {
    if (threadIdx.x < st) sm[threadIdx.x] += sm[threadIdx.x + st];
    __syncthreads();
  }
  if (threadIdx.x == 0) atomicAdd(out, sm[0]);
}

__global__ void k_finalize(const float* __restrict__ f, const unsigned* __restrict__ u,
                           float* __restrict__ out) {
  if (threadIdx.x == 0 && blockIdx.x == 0) {
    const float inv_bd = 1.0f / (float)(BB * DD);
    float ml2 = 0.25f * (f[2] + f[3] + f[4] + f[5]) * inv_bd;
    float l1 = 1.0e-3f * f[0] / (float)BB;
    float aux = u[6] ? (1.0e-2f * f[6] * inv_bd) : 0.0f;
    out[0] = ml2 + l1 + aux;   // total_loss
    out[1] = ml2;              // mean_l2_loss
    out[2] = l1;               // l1_loss
    out[3] = aux;              // aux_loss
    out[4] = f[1] / (float)BB; // l0_norm
  }
}

// ---------------- launch ----------------
extern "C" void kernel_launch(void* const* d_in, const int* in_sizes, int n_in,
                              void* d_out, int out_size, void* d_ws, size_t ws_size,
                              hipStream_t stream) {
  (void)in_sizes; (void)n_in; (void)out_size; (void)ws_size;
  const float* x     = (const float*)d_in[0];
  const float* b_dec = (const float*)d_in[1];
  const float* b_enc = (const float*)d_in[2];
  const float* W_enc = (const float*)d_in[3];  // [D,S]
  const float* W_dec = (const float*)d_in[4];  // [S,D]
  const int*   nba   = (const int*)d_in[5];

  float* out      = (float*)d_out;
  float* out_topk = out + 5;
  float* recon    = out + 5 + N_TOTAL;

  char* ws = (char*)d_ws;
  size_t off = 0;
  auto alloc = [&](size_t bytes) { size_t o = off; off += (bytes + 255) & ~(size_t)255; return o; };
  bf16_t* wenc_t = (bf16_t*)(ws + alloc((size_t)SS * DD * 2));  // [S,D]
  bf16_t* wdec_t = (bf16_t*)(ws + alloc((size_t)DD * SS * 2));  // [D,S]
  bf16_t* xc     = (bf16_t*)(ws + alloc((size_t)BB * DD * 2));
  float*  acts   = (float*) (ws + alloc(N_TOTAL * 4));
  bf16_t* atopk  = (bf16_t*)(ws + alloc(N_TOTAL * 2));
  bf16_t* aaux   = (bf16_t*)(ws + alloc(N_TOTAL * 2));
  unsigned char* dead = (unsigned char*)(ws + alloc(SS));
  size_t zoff = alloc(256 + 256 + 4 * 4096 * 4);
  float*    fscal = (float*)(ws + zoff);            // 0:l1 1:nnz 2..5:l2 6:auxsq 7:thr 8:thr_aux
  unsigned* uscal = (unsigned*)(ws + zoff + 256);   // 0..2 sel_main, 3..5 sel_aux, 6 any_dead
  unsigned* h1m = (unsigned*)(ws + zoff + 512);
  unsigned* h2m = h1m + 4096;
  unsigned* h1a = h2m + 4096;
  unsigned* h2a = h1a + 4096;
  const int zero_words = (256 + 256 + 4 * 4096 * 4) / 4;

  k_zero_u32<<<128, 256, 0, stream>>>((unsigned*)(ws + zoff), zero_words);
  k_transpose_cvt<<<dim3(SS / 32, DD / 32), 256, 0, stream>>>(W_enc, wenc_t, DD, SS);
  k_transpose_cvt<<<dim3(DD / 32, SS / 32), 256, 0, stream>>>(W_dec, wdec_t, SS, DD);
  k_xc<<<(BB * DD) / 256, 256, 0, stream>>>(x, b_dec, xc);
  k_dead<<<SS / 256, 256, 0, stream>>>(nba, dead, &uscal[6]);
  k_init_recon<<<(BB * DD) / 256, 256, 0, stream>>>(b_dec, recon);

  // encoder
  k_encoder<<<dim3(SS / 512, BB / 64), 256, 0, stream>>>(xc, wenc_t, b_enc, acts);

  // global top-k threshold (2-level radix select over float bits)
  k_hist1<<<2048, 256, 0, stream>>>(acts, N_TOTAL, nullptr, h1m);
  k_pick1<<<1, 32, 0, stream>>>(h1m, TOPK_TOTAL, &uscal[0]);
  k_hist2<<<2048, 256, 0, stream>>>(acts, N_TOTAL, nullptr, &uscal[0], h2m);
  k_pick2<<<1, 32, 0, stream>>>(h2m, &uscal[0], &fscal[7]);
  k_scatter_topk<<<2048, 256, 0, stream>>>(acts, &fscal[7], out_topk, atopk, N_TOTAL, fscal);

  // hierarchical group decode + per-group L2
  const int gb[5] = {0, 2048, 6144, 14336, 30720};
  for (int g = 0; g < 4; ++g) {
    k_decode<<<dim3(DD / 512, BB / 64), 256, 0, stream>>>(atopk, wdec_t, recon, gb[g], gb[g + 1]);
    k_sqerr<<<1024, 256, 0, stream>>>(recon, x, BB * DD, &fscal[2 + g]);
  }

  // aux dead-feature path
  k_hist1<<<2048, 256, 0, stream>>>(acts, N_TOTAL, dead, h1a);
  k_pick1<<<1, 32, 0, stream>>>(h1a, KAUX_TOTAL, &uscal[3]);
  k_hist2<<<2048, 256, 0, stream>>>(acts, N_TOTAL, dead, &uscal[3], h2a);
  k_pick2<<<1, 32, 0, stream>>>(h2a, &uscal[3], &fscal[8]);
  k_scatter_aux<<<2048, 256, 0, stream>>>(acts, &fscal[8], dead, aaux, N_TOTAL);
  k_aux_gemm<<<dim3(DD / 512, BB / 64), 256, 0, stream>>>(aaux, wdec_t, recon, x, &fscal[6]);

  k_finalize<<<1, 32, 0, stream>>>(fscal, uscal, out);
}
